// MSHABlock_90082644066403
// MI455X (gfx1250) — compile-verified, matrix operand order checked by reading, not run on hardware
//
#include <hip/hip_runtime.h>
#include <hip/hip_bf16.h>

typedef __attribute__((ext_vector_type(16))) _Float16 v16h;
typedef __attribute__((ext_vector_type(8)))  _Float16 v8h;
typedef __attribute__((ext_vector_type(8)))  float    v8f;
typedef __attribute__((ext_vector_type(4)))  unsigned int u32x4;
typedef __attribute__((ext_vector_type(4)))  int      i32x4;
typedef __attribute__((ext_vector_type(8)))  int      i32x8;

#define HEADS   6
#define HD      32
#define CCH     192
#define EMB     192
#define NTOK    784
#define BATCH   32
#define HP      28
#define M_TOTAL (BATCH * NTOK)   // 25088
#define QKV_N   576
#define OUT_N   384
#define SCALE   0.17677669529663687f   // 1/sqrt(32)

// ---------------------------------------------------------------------------
// Fragment loader for 16-bit WMMA A-frags (16x32, row-major source) and
// B-frags when the B matrix is stored transposed row-major (N x K).
// ISA layout: lane L: row = L&15; elements 0..7 -> K = kb..kb+7,
// elements 8..15 -> K = 16+kb..16+kb+7, kb = (L>=16) ? 8 : 0.
// ---------------------------------------------------------------------------
__device__ __forceinline__ v16h load_frag(const _Float16* __restrict__ base,
                                          int row, int stride, int lane) {
    int kb = (lane & 16) ? 8 : 0;
    const _Float16* p = base + (size_t)row * stride + kb;
    v16h f;
    *reinterpret_cast<v8h*>(&f)       = *reinterpret_cast<const v8h*>(p);
    *(reinterpret_cast<v8h*>(&f) + 1) = *reinterpret_cast<const v8h*>(p + 16);
    return f;
}

// ---------------------------------------------------------------------------
// Tensor Data Mover: DMA a 2D f16 tile (tile_d0 x tile_d1, row pitch
// row_stride elems) from global into LDS at lds_byte_off (contiguous,
// x-fastest -> matches ldsB[row*tile_d0 + k]). D# packing per CDNA5 ISA
// CDNA5 ISA. Issued once per wave; TENSORcnt-tracked.
// ---------------------------------------------------------------------------
__device__ __forceinline__ void tdm_load_2d_f16(unsigned lds_byte_off,
                                                const void* gaddr,
                                                unsigned tensor_d0, unsigned tensor_d1,
                                                unsigned tile_d0, unsigned tile_d1,
                                                unsigned row_stride) {
    unsigned long long ga = (unsigned long long)(uintptr_t)gaddr;
    u32x4 g0;
    g0[0] = 1u;                                            // count=1, user D#
    g0[1] = lds_byte_off;                                  // lds_addr
    g0[2] = (unsigned)(ga & 0xFFFFFFFFu);                  // global_addr[31:0]
    g0[3] = (unsigned)((ga >> 32) & 0x01FFFFFFu)           // global_addr[56:32]
          | (2u << 30);                                    // type=2 ("image")
    i32x8 g1;
    g1[0] = (int)(1u << 16);                               // data_size=1 (2B), mask=0
    g1[1] = (int)((tensor_d0 & 0xFFFFu) << 16);            // tensor_dim0[15:0]
    g1[2] = (int)(((tensor_d0 >> 16) & 0xFFFFu)            // tensor_dim0[31:16]
          | ((tensor_d1 & 0xFFFFu) << 16));                // tensor_dim1[15:0]
    g1[3] = (int)(((tensor_d1 >> 16) & 0xFFFFu)            // tensor_dim1[31:16]
          | ((tile_d0 & 0xFFFFu) << 16));                  // tile_dim0
    g1[4] = (int)(tile_d1 & 0xFFFFu);                      // tile_dim1 (tile_dim2=0)
    g1[5] = (int)row_stride;                               // tensor_dim0_stride[31:0]
    g1[6] = 0;                                             // stride hi / dim1_stride lo
    g1[7] = 0;                                             // tensor_dim1_stride hi
    i32x4 z4 = {0, 0, 0, 0};
#if defined(__clang_major__) && (__clang_major__ >= 23)
    i32x8 z8 = {0, 0, 0, 0, 0, 0, 0, 0};
    __builtin_amdgcn_tensor_load_to_lds(g0, g1, z4, z4, z8, 0);
#else
    __builtin_amdgcn_tensor_load_to_lds(g0, g1, z4, z4, 0);
#endif
}

// Low 32 bits of a flat shared-aperture address are the wave-relative LDS
// address (ISA §10.2).
__device__ __forceinline__ unsigned lds_addr_of(const void* p) {
    return (unsigned)(uintptr_t)p;
}

// ---------------------------------------------------------------------------
// Kernel 0: transpose f32 weights -> f16 (N x K row-major).
// ---------------------------------------------------------------------------
__global__ void transpose_to_f16(const float* __restrict__ src,
                                 _Float16* __restrict__ dst,
                                 int rowsK, int colsN) {
    int idx = blockIdx.x * blockDim.x + threadIdx.x;
    if (idx >= rowsK * colsN) return;
    int n = idx / rowsK;
    int k = idx - n * rowsK;
    dst[idx] = (_Float16)src[(size_t)k * colsN + n];
}

// ---------------------------------------------------------------------------
// Kernel 1: LayerNorm over C=192 + 2x2 maxpool. One wave per pooled token.
// ---------------------------------------------------------------------------
__global__ void ln_pool_kernel(const float* __restrict__ x,
                               const float* __restrict__ gamma,
                               const float* __restrict__ beta,
                               _Float16* __restrict__ xp) {
    int lane = threadIdx.x & 31;
    int wave = threadIdx.x >> 5;
    int tok  = blockIdx.x * 8 + wave;          // [0, 25088)
    int b = tok / NTOK;
    int n = tok - b * NTOK;
    int ph = n / HP, pw = n - ph * HP;

    float g[6], be[6], mx[6];
#pragma unroll
    for (int j = 0; j < 6; ++j) {
        int c = lane + 32 * j;
        g[j] = gamma[c]; be[j] = beta[c]; mx[j] = -3.4e38f;
    }
#pragma unroll
    for (int dy = 0; dy < 2; ++dy)
#pragma unroll
    for (int dx = 0; dx < 2; ++dx) {
        const float* src = x + (((size_t)b * 56 + (ph * 2 + dy)) * 56 + (pw * 2 + dx)) * CCH;
        float v[6], s = 0.f, ss = 0.f;
#pragma unroll
        for (int j = 0; j < 6; ++j) {
            v[j] = src[lane + 32 * j];
            s += v[j]; ss += v[j] * v[j];
        }
#pragma unroll
        for (int m = 1; m < 32; m <<= 1) {
            s  += __shfl_xor(s,  m, 32);
            ss += __shfl_xor(ss, m, 32);
        }
        float mean = s * (1.f / 192.f);
        float var  = ss * (1.f / 192.f) - mean * mean;
        float rstd = rsqrtf(var + 1e-5f);
#pragma unroll
        for (int j = 0; j < 6; ++j) {
            float t = (v[j] - mean) * rstd * g[j] + be[j];
            mx[j] = fmaxf(mx[j], t);
        }
    }
    _Float16* dst = xp + (size_t)tok * CCH;
#pragma unroll
    for (int j = 0; j < 6; ++j) dst[lane + 32 * j] = (_Float16)mx[j];
}

// ---------------------------------------------------------------------------
// Kernel 2: QKV GEMM (25088x192 @ 192x576). B tile staged via TDM
// (tensor_load_to_lds), A-frags straight from global, WMMA f16->f32.
// ---------------------------------------------------------------------------
__global__ void qkv_gemm_kernel(const _Float16* __restrict__ xp,
                                const _Float16* __restrict__ wT,
                                _Float16* __restrict__ qf,
                                _Float16* __restrict__ kf,
                                _Float16* __restrict__ vT) {
    __shared__ _Float16 ldsB[64 * 32];
    int lane  = threadIdx.x & 31;
    int wave  = threadIdx.x >> 5;
    int tBase = blockIdx.x * 64 + wave * 16;
    int cBase = blockIdx.y * 64;
    int n15   = lane & 15;

    v8f acc[4];
#pragma unroll
    for (int s = 0; s < 4; ++s)
#pragma unroll
        for (int e = 0; e < 8; ++e) acc[s][e] = 0.f;

    for (int kk = 0; kk < CCH; kk += 32) {
        // DMA 64x32 B tile (wT rows cBase..cBase+63, cols kk..kk+31) into LDS
        if (wave == 0) {
            tdm_load_2d_f16(lds_addr_of(&ldsB[0]),
                            wT + (size_t)cBase * CCH + kk,
                            CCH, QKV_N, 32, 64, CCH);
            __builtin_amdgcn_s_wait_tensorcnt(0);
        }
        if (kk + 32 < CCH)
            __builtin_prefetch(xp + (size_t)(tBase + n15) * CCH + kk + 32, 0, 1);
        __syncthreads();

        v16h a = load_frag(xp + (size_t)tBase * CCH + kk, n15, CCH, lane);
#pragma unroll
        for (int s = 0; s < 4; ++s) {
            v16h bf = load_frag(&ldsB[0], s * 16 + n15, 32, lane);
            acc[s] = __builtin_amdgcn_wmma_f32_16x16x32_f16(
                false, a, false, bf, (short)0, acc[s], false, false);
        }
        __syncthreads();
    }

    int half8 = (lane & 16) ? 8 : 0;
#pragma unroll
    for (int s = 0; s < 4; ++s) {
        int col = cBase + s * 16 + n15;       // region is wave-uniform per s
#pragma unroll
        for (int r = 0; r < 8; ++r) {
            int t = tBase + r + half8;
            float val = acc[s][r];
            if (col < 192) {                                   // Q (pre-scaled)
                qf[(size_t)t * EMB + col] = (_Float16)(val * SCALE);
            } else if (col < 384) {                            // K
                kf[(size_t)t * EMB + (col - 192)] = (_Float16)val;
            } else {                                           // V, transposed
                int j = col - 384, head = j >> 5, d = j & 31;
                int bb = t / NTOK, tok = t - bb * NTOK;
                vT[(((size_t)bb * HEADS + head) * HD + d) * NTOK + tok] = (_Float16)val;
            }
        }
    }
}

// ---------------------------------------------------------------------------
// Kernel 3: fused flash attention. One wave per (b, head, 16-query tile).
// Per 32-key chunk: 2 WMMA (QK^T) -> bias + online softmax -> P via LDS
// transpose -> 2 WMMA (PV). Tail chunk of 16 keys zero-pads P.
// ---------------------------------------------------------------------------
__global__ void attn_kernel(const _Float16* __restrict__ qf,
                            const _Float16* __restrict__ kf,
                            const _Float16* __restrict__ vT,
                            const float* __restrict__ pos,
                            _Float16* __restrict__ ao) {
    __shared__ _Float16 ldsP[16 * 32];
    int lane  = threadIdx.x & 31;
    int qTile = blockIdx.x;     // 49
    int head  = blockIdx.y;     // 6
    int b     = blockIdx.z;     // 32
    int qBase = qTile * 16;
    int half8 = (lane & 16) ? 8 : 0;
    int n15   = lane & 15;

    const _Float16* qp = qf + ((size_t)b * NTOK) * EMB + head * HD;
    const _Float16* kp = kf + ((size_t)b * NTOK) * EMB + head * HD;
    const _Float16* vp = vT + ((size_t)b * HEADS + head) * HD * NTOK;
    const float*    pt = pos + head * 3025;   // (2*28-1)^2

    v16h aQ = load_frag(qp + (size_t)qBase * EMB, n15, EMB, lane);

    float mrun[8], lrun[8];
    v8f o0, o1;
#pragma unroll
    for (int r = 0; r < 8; ++r) { mrun[r] = -3.4e38f; lrun[r] = 0.f; o0[r] = 0.f; o1[r] = 0.f; }

    int qh[8], qw[8];
#pragma unroll
    for (int r = 0; r < 8; ++r) {
        int qi = qBase + r + half8;
        qh[r] = qi / HP; qw[r] = qi - qh[r] * HP;
    }

    for (int kc = 0; kc < NTOK; kc += 32) {
        bool full = (kc + 32 <= NTOK);    // 784 = 24*32 + 16 -> last chunk is 16
        v8f s0, s1;
#pragma unroll
        for (int e = 0; e < 8; ++e) { s0[e] = 0.f; s1[e] = 0.f; }

        v16h bK0 = load_frag(kp + (size_t)kc * EMB, n15, EMB, lane);
        s0 = __builtin_amdgcn_wmma_f32_16x16x32_f16(false, aQ, false, bK0, (short)0, s0, false, false);
        if (full) {
            v16h bK1 = load_frag(kp + (size_t)(kc + 16) * EMB, n15, EMB, lane);
            s1 = __builtin_amdgcn_wmma_f32_16x16x32_f16(false, aQ, false, bK1, (short)0, s1, false, false);
        }

        // relative-position bias, computed on the fly
        int k0 = kc + n15;
        int kh0 = k0 / HP, kw0 = k0 - kh0 * HP;
        int k1 = kc + 16 + n15;
        int kh1 = k1 / HP, kw1 = k1 - kh1 * HP;   // unused when !full

        float rowm[8];
#pragma unroll
        for (int r = 0; r < 8; ++r) {
            s0[r] += pt[(qh[r] - kh0 + 27) * 55 + (qw[r] - kw0 + 27)];
            if (full) {
                s1[r] += pt[(qh[r] - kh1 + 27) * 55 + (qw[r] - kw1 + 27)];
                rowm[r] = fmaxf(s0[r], s1[r]);
            } else {
                rowm[r] = s0[r];
            }
        }
#pragma unroll
        for (int r = 0; r < 8; ++r) {
            float v = rowm[r];
            v = fmaxf(v, __shfl_xor(v, 1));
            v = fmaxf(v, __shfl_xor(v, 2));
            v = fmaxf(v, __shfl_xor(v, 4));
            v = fmaxf(v, __shfl_xor(v, 8));
            rowm[r] = v;   // per-row max within the owning 16-lane half
        }
#pragma unroll
        for (int r = 0; r < 8; ++r) {
            float mnew = fmaxf(mrun[r], rowm[r]);
            float f = __expf(mrun[r] - mnew);
            mrun[r] = mnew;
            o0[r] *= f; o1[r] *= f;
            float p0 = __expf(s0[r] - mnew);
            float p1 = full ? __expf(s1[r] - mnew) : 0.f;
            float rs = p0 + p1;
            rs += __shfl_xor(rs, 1);
            rs += __shfl_xor(rs, 2);
            rs += __shfl_xor(rs, 4);
            rs += __shfl_xor(rs, 8);
            lrun[r] = lrun[r] * f + rs;
            int m = r + half8;
            ldsP[m * 32 + n15]      = (_Float16)p0;   // C-layout -> row-major P
            ldsP[m * 32 + n15 + 16] = (_Float16)p1;
        }
        __syncthreads();   // single wave: orders LDS + compiler fence

        v16h aP  = load_frag(&ldsP[0], n15, 32, lane);
        v16h bV0 = load_frag(vp + kc, n15,      NTOK, lane);   // d = 0..15
        v16h bV1 = load_frag(vp + kc, 16 + n15, NTOK, lane);   // d = 16..31
        o0 = __builtin_amdgcn_wmma_f32_16x16x32_f16(false, aP, false, bV0, (short)0, o0, false, false);
        o1 = __builtin_amdgcn_wmma_f32_16x16x32_f16(false, aP, false, bV1, (short)0, o1, false, false);
        __syncthreads();
    }

#pragma unroll
    for (int r = 0; r < 8; ++r) {
        float inv = 1.f / lrun[r];
        int t = qBase + r + half8;
        _Float16* dst = ao + ((size_t)b * NTOK + t) * EMB + head * HD;
        dst[n15]      = (_Float16)(o0[r] * inv);
        dst[n15 + 16] = (_Float16)(o1[r] * inv);
    }
}

// ---------------------------------------------------------------------------
// Kernel 4: output GEMM (25088x192 @ 192x384) + bias, f32 out. TDM-staged B.
// ---------------------------------------------------------------------------
__global__ void out_gemm_kernel(const _Float16* __restrict__ ao,
                                const _Float16* __restrict__ wT,
                                const float* __restrict__ bias,
                                float* __restrict__ out) {
    __shared__ _Float16 ldsB[64 * 32];
    int lane  = threadIdx.x & 31;
    int wave  = threadIdx.x >> 5;
    int tBase = blockIdx.x * 64 + wave * 16;
    int cBase = blockIdx.y * 64;
    int n15   = lane & 15;

    v8f acc[4];
#pragma unroll
    for (int s = 0; s < 4; ++s)
#pragma unroll
        for (int e = 0; e < 8; ++e) acc[s][e] = 0.f;

    for (int kk = 0; kk < CCH; kk += 32) {
        if (wave == 0) {
            tdm_load_2d_f16(lds_addr_of(&ldsB[0]),
                            wT + (size_t)cBase * CCH + kk,
                            CCH, OUT_N, 32, 64, CCH);
            __builtin_amdgcn_s_wait_tensorcnt(0);
        }
        if (kk + 32 < CCH)
            __builtin_prefetch(ao + (size_t)(tBase + n15) * CCH + kk + 32, 0, 1);
        __syncthreads();

        v16h a = load_frag(ao + (size_t)tBase * CCH + kk, n15, CCH, lane);
#pragma unroll
        for (int s = 0; s < 4; ++s) {
            v16h bf = load_frag(&ldsB[0], s * 16 + n15, 32, lane);
            acc[s] = __builtin_amdgcn_wmma_f32_16x16x32_f16(
                false, a, false, bf, (short)0, acc[s], false, false);
        }
        __syncthreads();
    }

    int half8 = (lane & 16) ? 8 : 0;
#pragma unroll
    for (int s = 0; s < 4; ++s) {
        int col = cBase + s * 16 + n15;
        float bb = bias[col];
#pragma unroll
        for (int r = 0; r < 8; ++r) {
            int t = tBase + r + half8;
            out[(size_t)t * OUT_N + col] = acc[s][r] + bb;
        }
    }
}

// ---------------------------------------------------------------------------
extern "C" void kernel_launch(void* const* d_in, const int* in_sizes, int n_in,
                              void* d_out, int out_size, void* d_ws, size_t ws_size,
                              hipStream_t stream) {
    const float* x     = (const float*)d_in[0];
    const float* gamma = (const float*)d_in[1];
    const float* beta  = (const float*)d_in[2];
    const float* w_qkv = (const float*)d_in[3];
    const float* pos   = (const float*)d_in[4];
    const float* w_out = (const float*)d_in[5];
    const float* b_out = (const float*)d_in[6];
    float* out = (float*)d_out;

    char* ws = (char*)d_ws;
    size_t off = 0;
    auto alloc = [&](size_t bytes) -> char* {
        char* p = ws + off;
        off = (off + bytes + 255) & ~(size_t)255;
        return p;
    };
    _Float16* xp    = (_Float16*)alloc((size_t)M_TOTAL * CCH * 2);
    _Float16* qf    = (_Float16*)alloc((size_t)M_TOTAL * EMB * 2);
    _Float16* kf    = (_Float16*)alloc((size_t)M_TOTAL * EMB * 2);
    _Float16* vT    = (_Float16*)alloc((size_t)M_TOTAL * EMB * 2 + 256); // pad: tail V-frag overreads <32B
    _Float16* ao    = (_Float16*)alloc((size_t)M_TOTAL * EMB * 2);
    _Float16* wqkvT = (_Float16*)alloc((size_t)QKV_N * CCH * 2);
    _Float16* woutT = (_Float16*)alloc((size_t)OUT_N * CCH * 2);
    (void)ws_size; (void)in_sizes; (void)n_in; (void)out_size;

    transpose_to_f16<<<(QKV_N * CCH + 255) / 256, 256, 0, stream>>>(w_qkv, wqkvT, CCH, QKV_N);
    transpose_to_f16<<<(OUT_N * CCH + 255) / 256, 256, 0, stream>>>(w_out, woutT, CCH, OUT_N);
    ln_pool_kernel<<<M_TOTAL / 8, 256, 0, stream>>>(x, gamma, beta, xp);
    qkv_gemm_kernel<<<dim3(M_TOTAL / 64, QKV_N / 64), 128, 0, stream>>>(xp, wqkvT, qf, kf, vT);
    attn_kernel<<<dim3(49, HEADS, BATCH), 32, 0, stream>>>(qf, kf, vT, pos, ao);
    out_gemm_kernel<<<dim3(M_TOTAL / 64, OUT_N / 64), 128, 0, stream>>>(ao, woutT, b_out, out);
}
